// GRAM_84516366450884
// MI455X (gfx1250) — matrix operand order repaired and checked by reading, not compiled, not run on hardware
//
#include <hip/hip_runtime.h>

// ---------------- CDNA5 types ----------------
typedef __attribute__((ext_vector_type(16))) __bf16 v16bf;
typedef __attribute__((ext_vector_type(8)))  __bf16 v8bf;
typedef __attribute__((ext_vector_type(8)))  float  v8f;

// Problem constants
#define T_   50
#define B_   128
#define V_   8000
#define E_   128
#define H_   128
#define ATT_ 100
#define ATTP 112      // ATT padded to 7 N-tiles
#define C_   283
#define CP_  288      // C padded to 18 N-tiles
#define L_   5
#define ML_  (V_ * L_)   // 40000 MLP rows
#define G3_  384         // 3*H

// ---- manual bf16 conversions (RNE) + fast transcendentals ----
__device__ __forceinline__ unsigned short f2bf_bits(float f) {
    unsigned u = __builtin_bit_cast(unsigned, f);
    u += 0x7fffu + ((u >> 16) & 1u);
    return (unsigned short)(u >> 16);
}
__device__ __forceinline__ v16bf cat8(v8bf lo, v8bf hi) {
    return __builtin_shufflevector(lo, hi, 0,1,2,3,4,5,6,7,8,9,10,11,12,13,14,15);
}
__device__ __forceinline__ float bf2f(unsigned short s) {
    unsigned u = ((unsigned)s) << 16;
    return __builtin_bit_cast(float, u);
}
__device__ __forceinline__ float fsig(float x)  { return 1.0f / (1.0f + __expf(-x)); }
__device__ __forceinline__ float ftanh(float x) { return 1.0f - 2.0f / (1.0f + __expf(2.0f * x)); }

// load a WMMA A-fragment (16-bit layout) from a bf16 row pointer
__device__ __forceinline__ v16bf load_a(const unsigned short* row, int k0, int half) {
    v8bf lo = *(const v8bf*)(row + k0 + half * 16);
    v8bf hi = *(const v8bf*)(row + k0 + half * 16 + 8);
    return cat8(lo, hi);
}

// =====================================================================
// Kernel 0: one-time weight conversions to WMMA-friendly bf16 layouts
// =====================================================================
__global__ void convert_weights_kernel(const float* __restrict__ W_att,
                                       const float* __restrict__ W_out,
                                       const float* __restrict__ w_ih,
                                       const float* __restrict__ w_hh,
                                       unsigned short* __restrict__ WattT,
                                       unsigned short* __restrict__ WoutT,
                                       unsigned short* __restrict__ Wih_bf,
                                       unsigned short* __restrict__ Whh_bf) {
    const int tid = blockIdx.x * blockDim.x + threadIdx.x;
    const int stride = gridDim.x * blockDim.x;
    for (int i = tid; i < ATTP * 2 * E_; i += stride) {
        int n = i >> 8, k = i & 255;
        WattT[i] = (n < ATT_) ? f2bf_bits(W_att[k * ATT_ + n]) : (unsigned short)0;
    }
    for (int i = tid; i < CP_ * H_; i += stride) {
        int c = i >> 7, e = i & 127;
        WoutT[i] = (c < C_) ? f2bf_bits(W_out[e * C_ + c]) : (unsigned short)0;
    }
    for (int i = tid; i < 3 * H_ * E_; i += stride) Wih_bf[i] = f2bf_bits(w_ih[i]);
    for (int i = tid; i < 3 * H_ * H_; i += stride) Whh_bf[i] = f2bf_bits(w_hh[i]);
}

// =====================================================================
// Kernel 1a: pre[i] = sum_t tanh( att_in[i] . WattT[:,t] + b[t] ) * v[t]
//   A-fragments hoisted and reused across both N-tiles per wave.
// =====================================================================
__global__ void mlp_pre_kernel(const float* __restrict__ W_emb,
                               const unsigned short* __restrict__ WattT,
                               const float* __restrict__ b_att,
                               const float* __restrict__ v_att,
                               const int*   __restrict__ leaves,
                               const int*   __restrict__ ancestors,
                               float* __restrict__ pre) {
    __shared__ float pre_lds[16];
    const int m0   = blockIdx.x * 16;
    const int tid  = threadIdx.x;           // 128 threads, 4 waves
    const int wave = tid >> 5;
    const int lane = tid & 31;
    const int half = lane >> 4;
    const int l15  = lane & 15;

    if (tid < 16) pre_lds[tid] = 0.0f;

    const int row = m0 + l15;
    const float* leafRow = W_emb + (size_t)leaves[row]    * E_;
    const float* ancRow  = W_emb + (size_t)ancestors[row] * E_;

    // gather + convert A fragments once (K = 256 -> 8 fragments)
    v16bf a_frag[8];
#pragma unroll
    for (int ks = 0; ks < 8; ++ks) {
        const int k0 = ks * 32;
        const float* src = (k0 < 128) ? (leafRow + k0) : (ancRow + k0 - 128);
        v16bf a;
#pragma unroll
        for (int j = 0; j < 8; ++j) {
            a[j]     = __builtin_bit_cast(__bf16, f2bf_bits(src[half * 16 + j]));
            a[j + 8] = __builtin_bit_cast(__bf16, f2bf_bits(src[half * 16 + 8 + j]));
        }
        a_frag[ks] = a;
    }
    __syncthreads();

#pragma unroll
    for (int sub = 0; sub < 2; ++sub) {
        const int nt = wave * 2 + sub;      // 0..7 (tile 7 lies past ATTP rows; results discarded)
        const int n0 = nt * 16;
        const unsigned short* bcol = WattT + (size_t)(n0 + l15) * 256 + half * 16;
        v8f acc = {};
#pragma unroll
        for (int ks = 0; ks < 8; ++ks) {
            v16bf b = *(const v16bf*)(bcol + ks * 32);
            acc = __builtin_amdgcn_wmma_f32_16x16x32_bf16(
                false, a_frag[ks], false, b, (short)0, acc, false, false);
        }
        const int t = n0 + l15;
        if (t < ATT_) {
            const float bt = b_att[t], vt = v_att[t];
#pragma unroll
            for (int r = 0; r < 8; ++r)
                atomicAdd(&pre_lds[r + half * 8], ftanh(acc[r] + bt) * vt);
        }
    }
    __syncthreads();
    if (tid < 16) pre[m0 + tid] = pre_lds[tid];
}

// =====================================================================
// Kernel 1b: softmax over L ancestors + weighted ancestor sum -> embT bf16
// =====================================================================
__global__ void emb_kernel(const float* __restrict__ W_emb,
                           const int*   __restrict__ ancestors,
                           const float* __restrict__ pre,
                           unsigned short* __restrict__ embT) {
    __shared__ float att[L_];
    const int v = blockIdx.x;
    const int tid = threadIdx.x;            // 128 threads
    if (tid == 0) {
        float p[L_], mx = -3.4e38f;
        for (int l = 0; l < L_; ++l) { p[l] = pre[v * L_ + l]; mx = fmaxf(mx, p[l]); }
        float s = 0.0f;
        for (int l = 0; l < L_; ++l) { p[l] = __expf(p[l] - mx); s += p[l]; }
        const float inv = 1.0f / s;
        for (int l = 0; l < L_; ++l) att[l] = p[l] * inv;
    }
    __syncthreads();
    float e = 0.0f;
    for (int l = 0; l < L_; ++l)
        e += att[l] * W_emb[(size_t)ancestors[v * L_ + l] * E_ + tid];
    embT[(size_t)tid * V_ + v] = f2bf_bits(e);
}

// =====================================================================
// Kernel 2: x_emb = tanh(x[T*B, V] @ emb[V, E]) -> bf16
//   32 M-rows/block; A staged once/block into LDS (double buffered).
// =====================================================================
__global__ void gemm_xemb_kernel(const float* __restrict__ x,
                                 const unsigned short* __restrict__ embT,
                                 unsigned short* __restrict__ x_emb_bf) {
    __shared__ __align__(16) unsigned short As[2][32 * 32];
    const int m0   = blockIdx.x * 32;
    const int tid  = threadIdx.x;           // 256 threads, 8 waves
    const int wave = tid >> 5;
    const int lane = tid & 31;
    const int half = lane >> 4;
    const int l15  = lane & 15;
    const int n0   = wave * 16;

    const unsigned short* bcol = embT + (size_t)(n0 + l15) * V_ + half * 16;
    const float* xbase = x + (size_t)m0 * V_;

    v8f acc0 = {}, acc1 = {};
#pragma unroll
    for (int q = 0; q < 4; ++q) {
        const int idx = tid + 256 * q;
        const int row = idx >> 5, k = idx & 31;
        As[0][idx] = f2bf_bits(xbase[(size_t)row * V_ + k]);
    }
    __syncthreads();

    for (int kb = 0; kb < V_ / 32; ++kb) {
        const int cur = kb & 1;
        if (kb + 1 < V_ / 32) {
            const float* src = xbase + (kb + 1) * 32;
#pragma unroll
            for (int q = 0; q < 4; ++q) {
                const int idx = tid + 256 * q;
                const int row = idx >> 5, k = idx & 31;
                __builtin_prefetch(src + (size_t)row * V_ + k + 64, 0, 0);
                As[cur ^ 1][idx] = f2bf_bits(src[(size_t)row * V_ + k]);
            }
        }
        v16bf b = *(const v16bf*)(bcol + kb * 32);

        acc0 = __builtin_amdgcn_wmma_f32_16x16x32_bf16(
            false, load_a(&As[cur][l15 * 32], 0, half), false, b, (short)0, acc0, false, false);
        acc1 = __builtin_amdgcn_wmma_f32_16x16x32_bf16(
            false, load_a(&As[cur][(16 + l15) * 32], 0, half), false, b, (short)0, acc1, false, false);
        __syncthreads();
    }
#pragma unroll
    for (int r = 0; r < 8; ++r) {
        const int col = n0 + l15;
        x_emb_bf[(size_t)(m0 + r + half * 8) * E_ + col]      = f2bf_bits(ftanh(acc0[r]));
        x_emb_bf[(size_t)(m0 + 16 + r + half * 8) * E_ + col] = f2bf_bits(ftanh(acc1[r]));
    }
}

// =====================================================================
// Kernel 3a: gx = x_emb @ W_ih^T + b_ih  (time-parallel GRU input GEMM)
//   16 M-rows/block, 24 N-tiles over 8 waves, A-fragments hoisted.
// =====================================================================
__global__ void gx_kernel(const unsigned short* __restrict__ x_emb_bf,
                          const unsigned short* __restrict__ Wih_bf,
                          const float* __restrict__ b_ih,
                          float* __restrict__ gx) {
    const int m0   = blockIdx.x * 16;
    const int tid  = threadIdx.x;           // 256 threads, 8 waves
    const int wave = tid >> 5;
    const int lane = tid & 31;
    const int half = lane >> 4;
    const int l15  = lane & 15;

    const unsigned short* arow = x_emb_bf + (size_t)(m0 + l15) * E_;
    v16bf afrag[4];
#pragma unroll
    for (int ks = 0; ks < 4; ++ks) afrag[ks] = load_a(arow, ks * 32, half);

#pragma unroll
    for (int s = 0; s < 3; ++s) {
        const int nt = wave + 8 * s;        // 0..23
        const int bj = nt * 16 + l15;
        const unsigned short* bcol = Wih_bf + (size_t)bj * E_ + half * 16;
        v8f acc = {};
#pragma unroll
        for (int ks = 0; ks < 4; ++ks) {
            v16bf b = *(const v16bf*)(bcol + ks * 32);
            acc = __builtin_amdgcn_wmma_f32_16x16x32_bf16(
                false, afrag[ks], false, b, (short)0, acc, false, false);
        }
        const float bi = b_ih[bj];
#pragma unroll
        for (int r = 0; r < 8; ++r)
            gx[(size_t)(m0 + r + half * 8) * G3_ + bj] = acc[r] + bi;
    }
}

// =====================================================================
// Kernel 3b: GRU recurrence only (h @ W_hh^T + gates).
//   8 workgroups, each owns 16 batch rows (independent recurrences).
//   W_hh fragments + biases register-resident across all 50 steps.
// =====================================================================
__global__ void gru_kernel(const float* __restrict__ gx,
                           const unsigned short* __restrict__ Whh_bf,
                           const float* __restrict__ b_hh,
                           unsigned short* __restrict__ hidden_bf) {
    __shared__ float          Gh[16 * G3_];   // 24576 B
    __shared__ unsigned short h_bf[16 * H_];  //  4096 B

    const int m0   = blockIdx.x * 16;         // batch slice
    const int tid  = threadIdx.x;             // 256 threads, 8 waves
    const int wave = tid >> 5;
    const int lane = tid & 31;
    const int half = lane >> 4;
    const int l15  = lane & 15;

    // preload loop-invariant B fragments (3 N-tiles per wave x K=128)
    v16bf bfrag[3][4];
    float bias[3];
    int   bjv[3];
#pragma unroll
    for (int s = 0; s < 3; ++s) {
        const int nt = wave + 8 * s;
        const int bj = nt * 16 + l15;
        bjv[s]  = bj;
        bias[s] = b_hh[bj];
        const unsigned short* bcol = Whh_bf + (size_t)bj * H_ + half * 16;
#pragma unroll
        for (int ks = 0; ks < 4; ++ks)
            bfrag[s][ks] = *(const v16bf*)(bcol + ks * 32);
    }

    for (int i = tid; i < 16 * H_; i += 256) h_bf[i] = 0;
    __syncthreads();

    for (int t = 0; t < T_; ++t) {
        // A fragments from LDS h (same for all 3 N-tiles of this wave)
        const unsigned short* hrow = h_bf + l15 * H_;
        v16bf afrag[4];
#pragma unroll
        for (int ks = 0; ks < 4; ++ks) afrag[ks] = load_a(hrow, ks * 32, half);

#pragma unroll
        for (int s = 0; s < 3; ++s) {
            v8f acc = {};
#pragma unroll
            for (int ks = 0; ks < 4; ++ks)
                acc = __builtin_amdgcn_wmma_f32_16x16x32_bf16(
                    false, afrag[ks], false, bfrag[s][ks], (short)0, acc, false, false);
#pragma unroll
            for (int r = 0; r < 8; ++r)
                Gh[(r + half * 8) * G3_ + bjv[s]] = acc[r] + bias[s];
        }
        __syncthreads();

        // gates + state update (16 x 128 elements over 256 threads)
        for (int idx = tid; idx < 16 * H_; idx += 256) {
            const int rr = idx >> 7, e = idx & 127;
            // gx row index = t*B_ + m0 + rr  (time-dependent input gates)
            const float* grow  = gx + (size_t)((size_t)t * B_ + m0 + rr) * G3_;
            const float* ghrow = Gh + rr * G3_;
            const float r = fsig(grow[e] + ghrow[e]);
            const float z = fsig(grow[H_ + e] + ghrow[H_ + e]);
            const float n = ftanh(grow[2 * H_ + e] + r * ghrow[2 * H_ + e]);
            const float hp = bf2f(h_bf[idx]);
            const float hnew = (1.0f - z) * n + z * hp;
            const unsigned short hb = f2bf_bits(hnew);
            hidden_bf[(size_t)t * (B_ * H_) + (m0 + rr) * H_ + e] = hb;
            h_bf[idx] = hb;
        }
        __syncthreads();
    }
}

// =====================================================================
// Kernel 4: out = softmax(hidden @ W_out + b_out) * mask  via bf16 WMMA
// =====================================================================
__global__ void out_kernel(const unsigned short* __restrict__ hidden_bf,
                           const unsigned short* __restrict__ WoutT,
                           const float* __restrict__ b_out,
                           const float* __restrict__ mask,
                           float* __restrict__ out) {
    __shared__ float logits[16][CP_];
    __shared__ float rmax[16], rinv[16], rmsk[16];
    const int m0   = blockIdx.x * 16;
    const int tid  = threadIdx.x;           // 256 threads, 8 waves
    const int wave = tid >> 5;
    const int lane = tid & 31;
    const int half = lane >> 4;
    const int l15  = lane & 15;

    const unsigned short* arow = hidden_bf + (size_t)(m0 + l15) * H_;
    v16bf afrag[4];
#pragma unroll
    for (int ks = 0; ks < 4; ++ks) afrag[ks] = load_a(arow, ks * 32, half);

    for (int nt = wave; nt < CP_ / 16; nt += 8) {
        const int n0 = nt * 16;
        const unsigned short* bcol = WoutT + (size_t)(n0 + l15) * H_ + half * 16;
        v8f acc = {};
#pragma unroll
        for (int ks = 0; ks < 4; ++ks) {
            v16bf b = *(const v16bf*)(bcol + ks * 32);
            acc = __builtin_amdgcn_wmma_f32_16x16x32_bf16(
                false, afrag[ks], false, b, (short)0, acc, false, false);
        }
        const int c = n0 + l15;
        const float bo = (c < C_) ? b_out[c] : 0.0f;
#pragma unroll
        for (int r = 0; r < 8; ++r)
            logits[r + half * 8][c] = acc[r] + bo;
    }
    __syncthreads();

    if (tid < 16) {
        float mx = -3.4e38f;
        for (int c = 0; c < C_; ++c) mx = fmaxf(mx, logits[tid][c]);
        float s = 0.0f;
        for (int c = 0; c < C_; ++c) s += __expf(logits[tid][c] - mx);
        rmax[tid] = mx;
        rinv[tid] = 1.0f / s;
        rmsk[tid] = mask[m0 + tid];
    }
    __syncthreads();

    for (int idx = tid; idx < 16 * C_; idx += 256) {
        const int r = idx / C_, c = idx - r * C_;
        out[(size_t)(m0 + r) * C_ + c] =
            __expf(logits[r][c] - rmax[r]) * rinv[r] * rmsk[r];
    }
}

// =====================================================================
// Launch
// =====================================================================
extern "C" void kernel_launch(void* const* d_in, const int* in_sizes, int n_in,
                              void* d_out, int out_size, void* d_ws, size_t ws_size,
                              hipStream_t stream) {
    const float* x      = (const float*)d_in[0];
    const float* mask   = (const float*)d_in[1];
    const float* W_emb  = (const float*)d_in[2];
    const float* W_att  = (const float*)d_in[3];
    const float* b_att  = (const float*)d_in[4];
    const float* v_att  = (const float*)d_in[5];
    const float* w_ih   = (const float*)d_in[6];
    const float* w_hh   = (const float*)d_in[7];
    const float* b_ih   = (const float*)d_in[8];
    const float* b_hh   = (const float*)d_in[9];
    const float* W_out  = (const float*)d_in[10];
    const float* b_out  = (const float*)d_in[11];
    const int*   leaves    = (const int*)d_in[12];
    const int*   ancestors = (const int*)d_in[13];

    char* ws = (char*)d_ws;
    unsigned short* embT      = (unsigned short*)(ws + 0);         // 2,048,000 B
    unsigned short* x_emb_bf  = (unsigned short*)(ws + 2048000);   // 1,638,400 B
    unsigned short* hidden_bf = (unsigned short*)(ws + 3686400);   // 1,638,400 B
    float*          pre       = (float*)(ws + 5324800);            //   160,000 B
    unsigned short* WattT     = (unsigned short*)(ws + 5484800);   //    57,344 B
    unsigned short* WoutT     = (unsigned short*)(ws + 5542144);   //    73,728 B
    unsigned short* Wih_bf    = (unsigned short*)(ws + 5615872);   //    98,304 B
    unsigned short* Whh_bf    = (unsigned short*)(ws + 5714176);   //    98,304 B
    float*          gx        = (float*)(ws + 5812480);            // 9,830,400 B
    float* out = (float*)d_out;

    convert_weights_kernel<<<64, 256, 0, stream>>>(W_att, W_out, w_ih, w_hh,
                                                   WattT, WoutT, Wih_bf, Whh_bf);
    mlp_pre_kernel<<<ML_ / 16, 128, 0, stream>>>(W_emb, WattT, b_att, v_att,
                                                 leaves, ancestors, pre);
    emb_kernel<<<V_, 128, 0, stream>>>(W_emb, ancestors, pre, embT);
    gemm_xemb_kernel<<<(T_ * B_) / 32, 256, 0, stream>>>(x, embT, x_emb_bf);
    gx_kernel<<<(T_ * B_) / 16, 256, 0, stream>>>(x_emb_bf, Wih_bf, b_ih, gx);
    gru_kernel<<<B_ / 16, 256, 0, stream>>>(gx, Whh_bf, b_hh, hidden_bf);
    out_kernel<<<(T_ * B_) / 16, 256, 0, stream>>>(hidden_bf, WoutT, b_out, mask, out);
}
